// RBFKernelLayer_2568390443649
// MI455X (gfx1250) — matrix operand order repaired and checked by reading, not compiled, not run on hardware
//
#include <hip/hip_runtime.h>
#include <hip/hip_bf16.h>

// CDNA5 / gfx1250 RBF kernel layer:
//   K = exp(-max(||a||^2 + ||b||^2 - 2 a.b, 0) / (2*sigma^2))
//
// Pass 1 (prep, ~15MB traffic): f32 -> bf16 copies of A,B into d_ws and
//   row norms (80KB).
// Pass 2 (hot): 128x128 tile GEMM on V_WMMA_F32_16X16X32_BF16 with f32
//   accumulate, epilogue folded to add+fma+exp2+min per output:
//     x = c2*nA + c2*nB - 2*c2*(a.b),  c2 = -log2(e)/(2 sig^2)  (c2 < 0)
//     out = min(exp2(x), 1)   // min-with-1 == max(d2,0) clamp since c2<0
// Output-bandwidth bound: 256MB stores -> ~11us floor @ 23.3 TB/s.

typedef __attribute__((ext_vector_type(16))) __bf16 v16bf;
typedef __attribute__((ext_vector_type(8)))  __bf16 v8bf;
typedef __attribute__((ext_vector_type(16))) float  v16f;
typedef __attribute__((ext_vector_type(8)))  float  v8f;
typedef __attribute__((ext_vector_type(4)))  float  v4f;

#define D 128  // feature dim (GEMM K), fixed by the reference
#define LOG2E 1.44269504088896340736f

// ---------------------------------------------------------------- prep pass
// 8 threads per row; each converts 16 floats and accumulates squares.
__global__ __launch_bounds__(256) void prep_kernel(
    const float* __restrict__ A, const float* __restrict__ B,
    __bf16* __restrict__ Abf, __bf16* __restrict__ Bbf,
    float* __restrict__ nA, float* __restrict__ nB, int M, int N) {
  const int gid = blockIdx.x * 256 + threadIdx.x;
  const int row = gid >> 3;
  const int sub = gid & 7;
  if (row >= M + N) return;

  const float* src;
  __bf16* dst;
  float* np;
  int r;
  if (row < M) {
    r = row;      src = A + (size_t)row * D; dst = Abf + (size_t)row * D; np = nA;
  } else {
    r = row - M;  src = B + (size_t)r   * D; dst = Bbf + (size_t)r   * D; np = nB;
  }

  const float* p = src + sub * 16;
  v4f x0 = *(const v4f*)(p);
  v4f x1 = *(const v4f*)(p + 4);
  v4f x2 = *(const v4f*)(p + 8);
  v4f x3 = *(const v4f*)(p + 12);

  float s = 0.f;
#pragma unroll
  for (int i = 0; i < 4; ++i)
    s += x0[i]*x0[i] + x1[i]*x1[i] + x2[i]*x2[i] + x3[i]*x3[i];
  // reduce across the 8 lanes owning this row (lanes are contiguous)
  s += __shfl_xor(s, 1, 32);
  s += __shfl_xor(s, 2, 32);
  s += __shfl_xor(s, 4, 32);

  v8f lo  = __builtin_shufflevector(x0, x1, 0,1,2,3,4,5,6,7);
  v8f hi  = __builtin_shufflevector(x2, x3, 0,1,2,3,4,5,6,7);
  v16f xf = __builtin_shufflevector(lo, hi, 0,1,2,3,4,5,6,7,8,9,10,11,12,13,14,15);
  v16bf b = __builtin_convertvector(xf, v16bf);
  *(v16bf*)(dst + sub * 16) = b;   // 32B aligned

  if (sub == 0) np[r] = s;
}

// ---------------------------------------------------------------- hot pass
__global__ __launch_bounds__(256) void rbf_gemm_bf16(
    const __bf16* __restrict__ Abf,   // (M,128) bf16 row-major
    const __bf16* __restrict__ Bbf,   // (N,128) bf16 row-major
    const float* __restrict__ nA,     // (M) row norms
    const float* __restrict__ nB,     // (N) row norms
    const float* __restrict__ sigp,
    float* __restrict__ out, int ncols) {
  const int tid  = threadIdx.x;
  const int lane = tid & 31;
  const int wave = tid >> 5;
  const int wm   = wave & 3;    // 4 strips of 32 rows
  const int wn   = wave >> 2;   // 2 strips of 64 cols
  const int l15  = lane & 15;
  const int hi   = lane >> 4;   // 0 | 1

  const int rowBase = blockIdx.y * 128 + wm * 32;
  const int colBase = blockIdx.x * 128 + wn * 64;

  // Warm WGP$/L2 for the block's operand rows (global_prefetch_b8).
  __builtin_prefetch(Abf + (size_t)(blockIdx.y * 128 + (tid & 127)) * D, 0, 1);
  __builtin_prefetch(Bbf + (size_t)(blockIdx.x * 128 + (tid & 127)) * D, 0, 1);

  v8f acc[2][4];
#pragma unroll
  for (int t = 0; t < 2; ++t)
#pragma unroll
    for (int n = 0; n < 4; ++n)
      acc[t][n] = (v8f){0.f,0.f,0.f,0.f,0.f,0.f,0.f,0.f};

#pragma unroll
  for (int kb = 0; kb < D; kb += 32) {
    // A 16x32 bf16 fragment: lane<16 K={kb..kb+7, kb+16..23}; lane>=16 +8
    v16bf aF[2];
#pragma unroll
    for (int t = 0; t < 2; ++t) {
      const __bf16* p = Abf + (size_t)(rowBase + 16 * t + l15) * D + kb + hi * 8;
      v8bf a0 = *(const v8bf*)(p);
      v8bf a1 = *(const v8bf*)(p + 16);
      aF[t] = __builtin_shufflevector(a0, a1,
                                      0,1,2,3,4,5,6,7,8,9,10,11,12,13,14,15);
    }
    // B 32x16 bf16 fragment: lane<16 K={kb..kb+7, kb+16..23}; lane>=16 +8
    v16bf bF[4];
#pragma unroll
    for (int n = 0; n < 4; ++n) {
      const __bf16* q = Bbf + (size_t)(colBase + 16 * n + l15) * D + kb + hi * 8;
      v8bf b0 = *(const v8bf*)(q);
      v8bf b1 = *(const v8bf*)(q + 16);
      bF[n] = __builtin_shufflevector(b0, b1,
                                      0,1,2,3,4,5,6,7,8,9,10,11,12,13,14,15);
    }
#pragma unroll
    for (int t = 0; t < 2; ++t)
#pragma unroll
      for (int n = 0; n < 4; ++n)
        acc[t][n] = __builtin_amdgcn_wmma_f32_16x16x32_bf16(
            false, aF[t], false, bF[n], (short)0, acc[t][n], false, false);
  }

  const float sig = fmaxf(sigp[0], 0.001f);
  const float c2  = -0.5f / (sig * sig) * LOG2E;   // < 0
  const float a2c = -2.0f * c2;

  float cnB[4];
#pragma unroll
  for (int n = 0; n < 4; ++n)
    cnB[n] = c2 * nB[colBase + 16 * n + l15];      // lane's own column

#pragma unroll
  for (int t = 0; t < 2; ++t) {
    // lanes 0-15 need norms of rows base..base+7; lanes 16-31 base+8..+15
    const v8f rAv = *(const v8f*)(nA + rowBase + 16 * t + hi * 8);
    v8f crA = (v8f){c2,c2,c2,c2,c2,c2,c2,c2} * rAv;
#pragma unroll
    for (int n = 0; n < 4; ++n) {
      const size_t col = (size_t)(colBase + 16 * n + l15);
#pragma unroll
      for (int v = 0; v < 8; ++v) {
        const size_t row = (size_t)(rowBase + 16 * t + hi * 8 + v);
        const float x = fmaf(a2c, acc[t][n][v], crA[v] + cnB[n]);
        const float e = __builtin_amdgcn_exp2f(x);   // v_exp_f32
        out[row * (size_t)ncols + col] = fminf(e, 1.0f);
      }
    }
  }
}

// ------------------------------------------- fallback (no-workspace) kernel
__global__ __launch_bounds__(256) void rbf_wmma_inline(
    const float* __restrict__ A, const float* __restrict__ B,
    const float* __restrict__ sigp, float* __restrict__ out, int ncols) {
  const int tid  = threadIdx.x;
  const int lane = tid & 31;
  const int wave = tid >> 5;
  const int wm   = wave & 3;
  const int wn   = wave >> 2;
  const int l15  = lane & 15;
  const int hi   = lane >> 4;

  const int rowBase = blockIdx.y * 128 + wm * 32;
  const int colBase = blockIdx.x * 128 + wn * 64;

  v8f acc[2][4];
#pragma unroll
  for (int t = 0; t < 2; ++t)
#pragma unroll
    for (int n = 0; n < 4; ++n)
      acc[t][n] = (v8f){0.f,0.f,0.f,0.f,0.f,0.f,0.f,0.f};

  float saA[2] = {0.f, 0.f};
  float saB[4] = {0.f, 0.f, 0.f, 0.f};

#pragma unroll
  for (int kb = 0; kb < D; kb += 32) {
    v16bf aF[2], bF[4];
#pragma unroll
    for (int t = 0; t < 2; ++t) {
      const float* p = A + (size_t)(rowBase + 16 * t + l15) * D + kb + hi * 8;
      v4f x0 = *(const v4f*)(p);
      v4f x1 = *(const v4f*)(p + 4);
      v4f x2 = *(const v4f*)(p + 16);
      v4f x3 = *(const v4f*)(p + 20);
#pragma unroll
      for (int i = 0; i < 4; ++i)
        saA[t] += x0[i]*x0[i] + x1[i]*x1[i] + x2[i]*x2[i] + x3[i]*x3[i];
      v8f lo  = __builtin_shufflevector(x0, x1, 0,1,2,3,4,5,6,7);
      v8f hi2 = __builtin_shufflevector(x2, x3, 0,1,2,3,4,5,6,7);
      v16f xf = __builtin_shufflevector(lo, hi2,
                                        0,1,2,3,4,5,6,7,8,9,10,11,12,13,14,15);
      aF[t] = __builtin_convertvector(xf, v16bf);
    }
#pragma unroll
    for (int n = 0; n < 4; ++n) {
      const float* p = B + (size_t)(colBase + 16 * n + l15) * D + kb + hi * 8;
      v4f x0 = *(const v4f*)(p);
      v4f x1 = *(const v4f*)(p + 4);
      v4f x2 = *(const v4f*)(p + 16);
      v4f x3 = *(const v4f*)(p + 20);
#pragma unroll
      for (int i = 0; i < 4; ++i)
        saB[n] += x0[i]*x0[i] + x1[i]*x1[i] + x2[i]*x2[i] + x3[i]*x3[i];
      v8f lo  = __builtin_shufflevector(x0, x1, 0,1,2,3,4,5,6,7);
      v8f hi2 = __builtin_shufflevector(x2, x3, 0,1,2,3,4,5,6,7);
      v16f xf = __builtin_shufflevector(lo, hi2,
                                        0,1,2,3,4,5,6,7,8,9,10,11,12,13,14,15);
      bF[n] = __builtin_convertvector(xf, v16bf);
    }
#pragma unroll
    for (int t = 0; t < 2; ++t)
#pragma unroll
      for (int n = 0; n < 4; ++n)
        acc[t][n] = __builtin_amdgcn_wmma_f32_16x16x32_bf16(
            false, aF[t], false, bF[n], (short)0, acc[t][n], false, false);
  }

  const float sig = fmaxf(sigp[0], 0.001f);
  const float c2  = -0.5f / (sig * sig) * LOG2E;
  const float a2c = -2.0f * c2;

  float cnB[4];
#pragma unroll
  for (int n = 0; n < 4; ++n)
    cnB[n] = c2 * (saB[n] + __shfl_xor(saB[n], 16, 32));

  float crA[2][8];
#pragma unroll
  for (int t = 0; t < 2; ++t) {
    const float full = saA[t] + __shfl_xor(saA[t], 16, 32);
#pragma unroll
    for (int v = 0; v < 8; ++v)
      crA[t][v] = c2 * __shfl(full, v + hi * 8, 32);
  }

#pragma unroll
  for (int t = 0; t < 2; ++t)
#pragma unroll
    for (int n = 0; n < 4; ++n) {
      const size_t col = (size_t)(colBase + 16 * n + l15);
#pragma unroll
      for (int v = 0; v < 8; ++v) {
        const size_t row = (size_t)(rowBase + 16 * t + hi * 8 + v);
        const float x = fmaf(a2c, acc[t][n][v], crA[t][v] + cnB[n]);
        out[row * (size_t)ncols + col] = fminf(__builtin_amdgcn_exp2f(x), 1.0f);
      }
    }
}

// ---------------------------------------------------------------- launcher
extern "C" void kernel_launch(void* const* d_in, const int* in_sizes, int n_in,
                              void* d_out, int out_size, void* d_ws, size_t ws_size,
                              hipStream_t stream) {
  (void)n_in; (void)out_size;
  const float* A   = (const float*)d_in[0];  // (8,2048,128) -> (16384,128)
  const float* B   = (const float*)d_in[1];  // (4096,128)
  const float* sig = (const float*)d_in[2];
  float* out = (float*)d_out;                // (16384,4096)

  const int M = in_sizes[0] / D;  // 16384
  const int N = in_sizes[1] / D;  // 4096

  const size_t abytes = (size_t)in_sizes[0] * sizeof(__bf16);
  const size_t bbytes = (size_t)in_sizes[1] * sizeof(__bf16);
  const size_t need   = abytes + bbytes + (size_t)(M + N) * sizeof(float);

  dim3 grid(N / 128, M / 128);

  if (ws_size >= need && d_ws != nullptr) {
    __bf16* Abf = (__bf16*)d_ws;
    __bf16* Bbf = (__bf16*)((char*)d_ws + abytes);
    float*  nAp = (float*)((char*)d_ws + abytes + bbytes);
    float*  nBp = nAp + M;

    const int rows   = M + N;
    const int blocks = (rows * 8 + 255) / 256;
    prep_kernel<<<blocks, 256, 0, stream>>>(A, B, Abf, Bbf, nAp, nBp, M, N);
    rbf_gemm_bf16<<<grid, dim3(256), 0, stream>>>(Abf, Bbf, nAp, nBp, sig, out, N);
  } else {
    rbf_wmma_inline<<<grid, dim3(256), 0, stream>>>(A, B, sig, out, N);
  }
}
